// Decoder_Block_28570122453041
// MI455X (gfx1250) — compile-verified, matrix operand order checked by reading, not run on hardware
//
#include <hip/hip_runtime.h>
#include <hip/hip_bf16.h>

#define DIM 128
#define NTOK 128
#define ST 136           // padded f16 row stride (272B, 16B-aligned rows)
#define EPS 1e-5f

typedef _Float16 h8   __attribute__((ext_vector_type(8)));
typedef _Float16 v16h __attribute__((ext_vector_type(16)));
typedef float    v8f  __attribute__((ext_vector_type(8)));

__device__ __forceinline__ v8f wmma16(v16h a, v16h b, v8f c) {
  // D = A(16x32 f16) x B(32x16 f16) + C(16x16 f32)
  return __builtin_amdgcn_wmma_f32_16x16x32_f16(false, a, false, b, (short)0, c,
                                                false, false);
}

// Load one A/B fragment from a row-major (A: M-major, B: N-major/pre-transposed)
// f16 buffer. lane<16 holds K {k0..k0+7, k0+16..k0+23}; lane>=16 holds +8.
__device__ __forceinline__ v16h load_frag(const _Float16* base, int rowbase,
                                          int stride, int k0, int lane) {
  int row = rowbase + (lane & 15);
  int ks  = k0 + ((lane >> 4) << 3);
  const _Float16* p = base + row * stride + ks;
  h8 lo = *(const h8*)(p);
  h8 hi = *(const h8*)(p + 16);
  v16h r;
#pragma unroll
  for (int t = 0; t < 8; ++t) { r[t] = lo[t]; r[t + 8] = hi[t]; }
  return r;
}

// Per-row LayerNorm of a 128x128 fp32 global tile into f16 LDS (stride ST).
__device__ __forceinline__ void ln_rows_to_lds(const float* rowbase, const float* g,
                                               const float* bia, _Float16* dst,
                                               int wave, int lane) {
  for (int r = wave; r < 128; r += 8) {
    const float* x = rowbase + (size_t)r * DIM;
    float v0[4];
    float s = 0.f;
#pragma unroll
    for (int t = 0; t < 4; ++t) { v0[t] = x[lane + 32 * t]; s += v0[t]; }
#pragma unroll
    for (int off = 16; off; off >>= 1) s += __shfl_xor(s, off, 32);
    float mu = s * (1.f / 128.f);
    float vs = 0.f;
#pragma unroll
    for (int t = 0; t < 4; ++t) { float d = v0[t] - mu; vs += d * d; }
#pragma unroll
    for (int off = 16; off; off >>= 1) vs += __shfl_xor(vs, off, 32);
    float inv = rsqrtf(vs * (1.f / 128.f) + EPS);
#pragma unroll
    for (int t = 0; t < 4; ++t) {
      int c = lane + 32 * t;
      dst[r * ST + c] = (_Float16)(((v0[t] - mu) * inv) * g[c] + bia[c]);
    }
  }
}

// In-register row LayerNorm over 8 accumulator tiles (one 16-row strip per wave).
// Row values for row (vgpr r, half) live in 16 lanes x 8 ct. Reduce with
// xor-shuffles (masks 1,2,4,8 stay within each 16-lane half). Store fp32 global.
__device__ __forceinline__ void rowln_store_global(v8f* accs, int jt, int lane,
                                                   const float* g, const float* b,
                                                   float* outb) {
#pragma unroll
  for (int r = 0; r < 8; ++r) {
    float s = 0.f;
#pragma unroll
    for (int ct = 0; ct < 8; ++ct) s += accs[ct][r];
#pragma unroll
    for (int off = 1; off < 16; off <<= 1) s += __shfl_xor(s, off, 32);
    float mu = s * (1.f / 128.f);
    float vs = 0.f;
#pragma unroll
    for (int ct = 0; ct < 8; ++ct) { float d = accs[ct][r] - mu; vs += d * d; }
#pragma unroll
    for (int off = 1; off < 16; off <<= 1) vs += __shfl_xor(vs, off, 32);
    float inv = rsqrtf(vs * (1.f / 128.f) + EPS);
    int j = jt * 16 + ((lane >= 16) ? 8 : 0) + r;
#pragma unroll
    for (int ct = 0; ct < 8; ++ct) {
      int c = ct * 16 + (lane & 15);
      outb[(size_t)j * DIM + c] = (accs[ct][r] - mu) * inv * g[c] + b[c];
    }
  }
}

// Same but store f16 into LDS (stride ST).
__device__ __forceinline__ void rowln_store_lds(v8f* accs, int jt, int lane,
                                                const float* g, const float* b,
                                                _Float16* dst) {
#pragma unroll
  for (int r = 0; r < 8; ++r) {
    float s = 0.f;
#pragma unroll
    for (int ct = 0; ct < 8; ++ct) s += accs[ct][r];
#pragma unroll
    for (int off = 1; off < 16; off <<= 1) s += __shfl_xor(s, off, 32);
    float mu = s * (1.f / 128.f);
    float vs = 0.f;
#pragma unroll
    for (int ct = 0; ct < 8; ++ct) { float d = accs[ct][r] - mu; vs += d * d; }
#pragma unroll
    for (int off = 1; off < 16; off <<= 1) vs += __shfl_xor(vs, off, 32);
    float inv = rsqrtf(vs * (1.f / 128.f) + EPS);
    int j = jt * 16 + ((lane >= 16) ? 8 : 0) + r;
#pragma unroll
    for (int ct = 0; ct < 8; ++ct) {
      int c = ct * 16 + (lane & 15);
      dst[j * ST + c] = (_Float16)((accs[ct][r] - mu) * inv * g[c] + b[c]);
    }
  }
}

// ---------------- small kernels -------------------------------------------

// Transpose a 128x128 fp32 weight to f16 N-major (WT[n][k] = W[k][n]).
__global__ void __launch_bounds__(256) prep_w(const float* __restrict__ W,
                                              _Float16* __restrict__ WT) {
  int idx = blockIdx.x * 256 + threadIdx.x;  // 0..16383
  int n = idx >> 7, k = idx & 127;
  WT[n * 128 + k] = (_Float16)W[k * 128 + n];
}

// Row LayerNorm for (rows,128) fp32: one block per row.
__global__ void __launch_bounds__(128)
node_ln(const float* __restrict__ x, const float* __restrict__ g,
        const float* __restrict__ b, float* __restrict__ y) {
  __shared__ float red[4];
  int r = blockIdx.x, t = threadIdx.x;
  float v = x[(size_t)r * DIM + t];
  float s = v;
#pragma unroll
  for (int off = 16; off; off >>= 1) s += __shfl_xor(s, off, 32);
  if ((t & 31) == 0) red[t >> 5] = s;
  __syncthreads();
  float mu = (red[0] + red[1] + red[2] + red[3]) * (1.f / 128.f);
  __syncthreads();
  float d = v - mu;
  float vs = d * d;
#pragma unroll
  for (int off = 16; off; off >>= 1) vs += __shfl_xor(vs, off, 32);
  if ((t & 31) == 0) red[t >> 5] = vs;
  __syncthreads();
  float var = (red[0] + red[1] + red[2] + red[3]) * (1.f / 128.f);
  y[(size_t)r * DIM + t] = d * rsqrtf(var + EPS) * g[t] + b[t];
}

// y[r] = x[r] @ W + bias (+res) (optional relu). One block per row, fp32.
__global__ void __launch_bounds__(128)
node_linear(const float* __restrict__ x, const float* __restrict__ W,
            const float* __restrict__ bias, const float* __restrict__ res,
            float* __restrict__ y, int relu) {
  __shared__ float xs[128];
  int r = blockIdx.x, n = threadIdx.x;
  xs[n] = x[(size_t)r * DIM + n];
  __syncthreads();
  float acc = bias[n];
#pragma unroll 8
  for (int k = 0; k < 128; ++k) acc += xs[k] * W[k * 128 + n];
  if (res) acc += res[(size_t)r * DIM + n];
  if (relu) acc = fmaxf(acc, 0.f);
  y[(size_t)r * DIM + n] = acc;
}

// ---------------- fused attn_new (prot side) ------------------------------
// Per (b,i) tile: ln1_pa -> E = X@We -> gate with q,k -> softmax/V reduce
// -> out_e GEMM -> +X residual -> ln2_pa -> d_out.  2 WMMA GEMMs.
__global__ void __launch_bounds__(256)
attn_new_fused(const float* __restrict__ prot_adj, const float* g1, const float* b1,
               const _Float16* __restrict__ WeT, const float* be,
               const _Float16* __restrict__ WoeT, const float* boe,
               const float* __restrict__ q2, const float* __restrict__ k2,
               const float* __restrict__ v2, const float* g2, const float* b2,
               float* __restrict__ out_padj, float* __restrict__ S_px) {
  extern __shared__ char smem_raw[];
  _Float16* Xh = (_Float16*)smem_raw;     // 128 x ST, ln1_pa(prot_adj row)
  _Float16* Th = Xh + 128 * ST;           // 128 x ST, gated attn
  int blk = blockIdx.x;
  int b = blk >> 7;
  int tid = threadIdx.x, wave = tid >> 5, lane = tid & 31;
  const float* rowbase = prot_adj + (size_t)blk * DIM * NTOK;

  ln_rows_to_lds(rowbase, g1, b1, Xh, wave, lane);
  __syncthreads();

  int jt = wave;  // 16-row strip per wave
  const float* qrow = q2 + (size_t)blk * DIM;
  const float* kb = k2 + (size_t)b * NTOK * DIM;
#pragma unroll 1
  for (int ct = 0; ct < 8; ++ct) {
    v8f acc = {0.f, 0.f, 0.f, 0.f, 0.f, 0.f, 0.f, 0.f};
#pragma unroll
    for (int kk = 0; kk < 4; ++kk) {
      v16h a  = load_frag(Xh, jt * 16, ST, kk * 32, lane);
      v16h bf = load_frag(WeT, ct * 16, DIM, kk * 32, lane);
      acc = wmma16(a, bf, acc);
    }
    int c = ct * 16 + (lane & 15);
    int rb = jt * 16 + ((lane >= 16) ? 8 : 0);
    float qv = qrow[c] * 0.25f;  // 1/sqrt(dk)=1/4
    float bec = be[c];
#pragma unroll
    for (int r = 0; r < 8; ++r) {
      float e = acc[r] + bec;
      float att = (qv * kb[(size_t)(rb + r) * DIM + c]) * (e + 1.f) * e;
      Th[(rb + r) * ST + c] = (_Float16)att;
    }
  }
  __syncthreads();

  if (tid < 128) {  // softmax over j + a*v reduce -> S_px (pre-out_n)
    int c = tid;
    float m = -3.0e38f;
    for (int j = 0; j < 128; ++j) m = fmaxf(m, (float)Th[j * ST + c]);
    float s = 0.f, ns = 0.f;
    const float* vb = v2 + (size_t)b * NTOK * DIM;
    for (int j = 0; j < 128; ++j) {
      float e = __expf((float)Th[j * ST + c] - m);
      s += e;
      ns += e * vb[(size_t)j * DIM + c];
    }
    S_px[(size_t)blk * DIM + c] = ns / s;
  }

  // out_e GEMM + residual X + ln2_pa -> store
  v8f accs[8];
#pragma unroll 1
  for (int ct = 0; ct < 8; ++ct) {
    v8f acc = {0.f, 0.f, 0.f, 0.f, 0.f, 0.f, 0.f, 0.f};
#pragma unroll
    for (int kk = 0; kk < 4; ++kk) {
      v16h a  = load_frag(Th, jt * 16, ST, kk * 32, lane);
      v16h bf = load_frag(WoeT, ct * 16, DIM, kk * 32, lane);
      acc = wmma16(a, bf, acc);
    }
    int c = ct * 16 + (lane & 15);
    int rb = jt * 16 + ((lane >= 16) ? 8 : 0);
    float bo = boe[c];
#pragma unroll
    for (int r = 0; r < 8; ++r) acc[r] += bo + (float)Xh[(rb + r) * ST + c];
    accs[ct] = acc;
  }
  rowln_store_global(accs, jt, lane, g2, b2, out_padj + (size_t)blk * DIM * NTOK);
}

// ---------------- fused dec_attn + edge MLP (mol side) --------------------
// Per (b,i) tile: ln1_ma -> prot_e/mol_e GEMMs -> gate -> softmax/V reduce
// -> out_ed GEMM +res -> ln3_ma -> fc1(relu) -> fc2 +res -> ln4_ma -> d_out.
// 5 WMMA GEMMs.
__global__ void __launch_bounds__(256)
dec_attn_fused(const float* __restrict__ mol_adj, const float* g1, const float* b1,
               const float* __restrict__ padj2,
               const _Float16* __restrict__ WkpaT, const float* bkpa,
               const _Float16* __restrict__ WvmaT, const float* bvma,
               const _Float16* __restrict__ WoedT, const float* boed,
               const float* __restrict__ qd, const float* __restrict__ kd,
               const float* __restrict__ vd, const float* g3, const float* b3,
               const _Float16* __restrict__ Wf1T, const float* bf1,
               const _Float16* __restrict__ Wf2T, const float* bf2,
               const float* g4, const float* b4, float* __restrict__ out_madj,
               float* __restrict__ S_mx) {
  extern __shared__ char smem_raw[];
  _Float16* Mh = (_Float16*)smem_raw;   // ln1_ma(mol_adj row); later fc1 act
  _Float16* Ph = Mh + 128 * ST;         // prot_adj_ln2 row; later ma2
  _Float16* Th = Ph + 128 * ST;         // gated attn
  int blk = blockIdx.x;
  int b = blk >> 7;
  int tid = threadIdx.x, wave = tid >> 5, lane = tid & 31;
  const float* mrow = mol_adj + (size_t)blk * DIM * NTOK;
  const float* prow = padj2 + (size_t)blk * DIM * NTOK;

  ln_rows_to_lds(mrow, g1, b1, Mh, wave, lane);
  for (int r = wave; r < 128; r += 8) {
#pragma unroll
    for (int t = 0; t < 4; ++t) {
      int c = lane + 32 * t;
      Ph[r * ST + c] = (_Float16)prow[(size_t)r * DIM + c];
    }
  }
  __syncthreads();

  int jt = wave;
  const float* qrow = qd + (size_t)blk * DIM;
  const float* kb = kd + (size_t)b * NTOK * DIM;
  const float inv_sc = 0.08838834764831845f;  // 1/sqrt(128)
#pragma unroll 1
  for (int ct = 0; ct < 8; ++ct) {
    v8f accP = {0.f, 0.f, 0.f, 0.f, 0.f, 0.f, 0.f, 0.f};
    v8f accM = {0.f, 0.f, 0.f, 0.f, 0.f, 0.f, 0.f, 0.f};
#pragma unroll
    for (int kk = 0; kk < 4; ++kk) {
      v16h ap = load_frag(Ph, jt * 16, ST, kk * 32, lane);
      v16h bp = load_frag(WkpaT, ct * 16, DIM, kk * 32, lane);
      accP = wmma16(ap, bp, accP);
      v16h am = load_frag(Mh, jt * 16, ST, kk * 32, lane);
      v16h bm = load_frag(WvmaT, ct * 16, DIM, kk * 32, lane);
      accM = wmma16(am, bm, accM);
    }
    int c = ct * 16 + (lane & 15);
    int rb = jt * 16 + ((lane >= 16) ? 8 : 0);
    float qv = qrow[c] * inv_sc;
    float bpc = bkpa[c], bmc = bvma[c];
#pragma unroll
    for (int r = 0; r < 8; ++r) {
      float pe = accP[r] + bpc;
      float me = accM[r] + bmc;
      float att = (qv * kb[(size_t)(rb + r) * DIM + c]) * (pe + 1.f) * me;
      Th[(rb + r) * ST + c] = (_Float16)att;
    }
  }
  __syncthreads();

  if (tid < 128) {  // softmax over j + a*v reduce -> S_mx (pre-out_nd)
    int c = tid;
    float m = -3.0e38f;
    for (int j = 0; j < 128; ++j) m = fmaxf(m, (float)Th[j * ST + c]);
    float s = 0.f, ns = 0.f;
    const float* vb = vd + (size_t)b * NTOK * DIM;
    for (int j = 0; j < 128; ++j) {
      float e = __expf((float)Th[j * ST + c] - m);
      s += e;
      ns += e * vb[(size_t)j * DIM + c];
    }
    S_mx[(size_t)blk * DIM + c] = ns / s;
  }

  // out_ed GEMM + residual Mh -> ln3_ma -> Ph (ma2, f16)
  v8f accs[8];
#pragma unroll 1
  for (int ct = 0; ct < 8; ++ct) {
    v8f acc = {0.f, 0.f, 0.f, 0.f, 0.f, 0.f, 0.f, 0.f};
#pragma unroll
    for (int kk = 0; kk < 4; ++kk) {
      v16h a  = load_frag(Th, jt * 16, ST, kk * 32, lane);
      v16h bf = load_frag(WoedT, ct * 16, DIM, kk * 32, lane);
      acc = wmma16(a, bf, acc);
    }
    int c = ct * 16 + (lane & 15);
    int rb = jt * 16 + ((lane >= 16) ? 8 : 0);
    float bo = boed[c];
#pragma unroll
    for (int r = 0; r < 8; ++r) acc[r] += bo + (float)Mh[(rb + r) * ST + c];
    accs[ct] = acc;
  }
  rowln_store_lds(accs, jt, lane, g3, b3, Ph);
  __syncthreads();

  // fc1 + relu -> Mh (strip-local: each wave only touches its own 16 rows)
#pragma unroll 1
  for (int ct = 0; ct < 8; ++ct) {
    v8f acc = {0.f, 0.f, 0.f, 0.f, 0.f, 0.f, 0.f, 0.f};
#pragma unroll
    for (int kk = 0; kk < 4; ++kk) {
      v16h a  = load_frag(Ph, jt * 16, ST, kk * 32, lane);
      v16h bf = load_frag(Wf1T, ct * 16, DIM, kk * 32, lane);
      acc = wmma16(a, bf, acc);
    }
    int c = ct * 16 + (lane & 15);
    int rb = jt * 16 + ((lane >= 16) ? 8 : 0);
    float bc = bf1[c];
#pragma unroll
    for (int r = 0; r < 8; ++r)
      Mh[(rb + r) * ST + c] = (_Float16)fmaxf(acc[r] + bc, 0.f);
  }
  __syncthreads();

  // fc2 + residual ma2 -> ln4_ma -> d_out mol_adj
#pragma unroll 1
  for (int ct = 0; ct < 8; ++ct) {
    v8f acc = {0.f, 0.f, 0.f, 0.f, 0.f, 0.f, 0.f, 0.f};
#pragma unroll
    for (int kk = 0; kk < 4; ++kk) {
      v16h a  = load_frag(Mh, jt * 16, ST, kk * 32, lane);
      v16h bf = load_frag(Wf2T, ct * 16, DIM, kk * 32, lane);
      acc = wmma16(a, bf, acc);
    }
    int c = ct * 16 + (lane & 15);
    int rb = jt * 16 + ((lane >= 16) ? 8 : 0);
    float bc = bf2[c];
#pragma unroll
    for (int r = 0; r < 8; ++r) acc[r] += bc + (float)Ph[(rb + r) * ST + c];
    accs[ct] = acc;
  }
  rowln_store_global(accs, jt, lane, g4, b4, out_madj + (size_t)blk * DIM * NTOK);
}

// ---------------- launch ---------------------------------------------------

extern "C" void kernel_launch(void* const* d_in, const int* in_sizes, int n_in,
                              void* d_out, int out_size, void* d_ws, size_t ws_size,
                              hipStream_t stream) {
  (void)in_sizes; (void)n_in; (void)out_size; (void)ws_size;
  auto W = [&](int i) { return (const float*)d_in[i]; };
  const float* mol_annot  = (const float*)d_in[0];
  const float* prot_annot = (const float*)d_in[1];
  const float* mol_adj    = (const float*)d_in[2];
  const float* prot_adj   = (const float*)d_in[3];

  // workspace layout
  char* ws = (char*)d_ws;
  _Float16* WeT   = (_Float16*)ws;          // attn2.e
  _Float16* WoeT  = WeT + 16384;            // attn2.out_e
  _Float16* WkpaT = WoeT + 16384;           // dec.k_pa
  _Float16* WvmaT = WkpaT + 16384;          // dec.v_ma
  _Float16* WoedT = WvmaT + 16384;          // dec.out_ed
  _Float16* Wf1T  = WoedT + 16384;          // mlp_ma.fc1
  _Float16* Wf2T  = Wf1T + 16384;           // mlp_ma.fc2
  float* fb  = (float*)(ws + 262144);
  float* pal = fb + 0 * 262144;   // ln1_px(prot_annot)
  float* q2  = fb + 1 * 262144;
  float* k2  = fb + 2 * 262144;
  float* v2  = fb + 3 * 262144;
  float* Spx = fb + 4 * 262144;
  float* mal = fb + 5 * 262144;   // ln1_mx(mol_annot)
  float* qd  = fb + 6 * 262144;
  float* kd  = fb + 7 * 262144;
  float* vd  = fb + 8 * 262144;
  float* Smx = fb + 9 * 262144;
  float* tA  = fb + 10 * 262144;
  float* tB  = fb + 11 * 262144;
  float* tC  = fb + 12 * 262144;

  float* out      = (float*)d_out;
  float* out_mx   = out;                 // mol_annot
  float* out_px   = out + 262144;        // prot_annot
  float* out_madj = out + 524288;        // mol_adj
  float* out_padj = out + 34078720;      // prot_adj

  // f16-transposed weights for WMMA GEMMs
  prep_w<<<64, 256, 0, stream>>>(W(10), WeT);
  prep_w<<<64, 256, 0, stream>>>(W(12), WoeT);
  prep_w<<<64, 256, 0, stream>>>(W(22), WkpaT);
  prep_w<<<64, 256, 0, stream>>>(W(24), WvmaT);
  prep_w<<<64, 256, 0, stream>>>(W(28), WoedT);
  prep_w<<<64, 256, 0, stream>>>(W(30), Wf1T);
  prep_w<<<64, 256, 0, stream>>>(W(32), Wf2T);

  // node path pre-attention
  node_ln<<<2048, 128, 0, stream>>>(prot_annot, W(44), W(45), pal);  // ln1_px
  node_ln<<<2048, 128, 0, stream>>>(mol_annot,  W(42), W(43), mal);  // ln1_mx
  node_linear<<<2048, 128, 0, stream>>>(pal, W(4), W(5), nullptr, q2, 0);
  node_linear<<<2048, 128, 0, stream>>>(pal, W(6), W(7), nullptr, k2, 0);
  node_linear<<<2048, 128, 0, stream>>>(pal, W(8), W(9), nullptr, v2, 0);

  size_t shA = (size_t)2 * 128 * ST * sizeof(_Float16);   // ~70 KB
  attn_new_fused<<<2048, 256, shA, stream>>>(prot_adj, W(40), W(41), WeT, W(11),
                                             WoeT, W(13), q2, k2, v2, W(46),
                                             W(47), out_padj, Spx);

  // prot node output: out_n + residual, ln2_px -> final prot_annot
  node_linear<<<2048, 128, 0, stream>>>(Spx, W(14), W(15), pal, tA, 0);
  node_ln<<<2048, 128, 0, stream>>>(tA, W(48), W(49), out_px);

  // dec q/k/v
  node_linear<<<2048, 128, 0, stream>>>(mal,    W(16), W(17), nullptr, qd, 0);
  node_linear<<<2048, 128, 0, stream>>>(out_px, W(18), W(19), nullptr, kd, 0);
  node_linear<<<2048, 128, 0, stream>>>(mal,    W(20), W(21), nullptr, vd, 0);

  size_t shB = (size_t)3 * 128 * ST * sizeof(_Float16);   // ~105 KB
  dec_attn_fused<<<2048, 256, shB, stream>>>(mol_adj, W(38), W(39), out_padj,
                                             WkpaT, W(23), WvmaT, W(25), WoedT,
                                             W(29), qd, kd, vd, W(50), W(51),
                                             Wf1T, W(31), Wf2T, W(33), W(54),
                                             W(55), out_madj, Smx);

  // mol node output: out_nd + residual, ln3_mx, mlp_mx, ln4_mx -> final mol_annot
  node_linear<<<2048, 128, 0, stream>>>(Smx, W(26), W(27), mal, tA, 0);
  node_ln<<<2048, 128, 0, stream>>>(tA, W(52), W(53), tB);
  node_linear<<<2048, 128, 0, stream>>>(tB, W(34), W(35), nullptr, tA, 1);
  node_linear<<<2048, 128, 0, stream>>>(tA, W(36), W(37), tB, tC, 0);
  node_ln<<<2048, 128, 0, stream>>>(tC, W(56), W(57), out_mx);
}
